// SynGCN_73057393705526
// MI455X (gfx1250) — matrix operand order repaired and checked by reading, not compiled, not run on hardware
//
#include <hip/hip_runtime.h>
#include <hip/hip_bf16.h>

typedef __attribute__((ext_vector_type(16))) _Float16 v16h;
typedef __attribute__((ext_vector_type(8)))  _Float16 v8h;
typedef __attribute__((ext_vector_type(8)))  float    v8f;

#define B_   128
#define S_   256
#define EMBD 300
#define EMBP 320
#define H_   256
#define G4   1024
#define N_   32768
#define NE_  32768
#define NC_  42

__device__ __forceinline__ float sigf(float x) { return 1.0f / (1.0f + __expf(-x)); }

// ---------------- conversion / init kernels ----------------

__global__ void k_convert_pad(const float* __restrict__ in, _Float16* __restrict__ out,
                              int rows, int cin, int cpad) {
  int idx = blockIdx.x * blockDim.x + threadIdx.x;
  int total = rows * cpad;
  if (idx >= total) return;
  int r = idx / cpad, c = idx % cpad;
  out[idx] = (c < cin) ? (_Float16)in[r * cin + c] : (_Float16)0.0f;
}

// out[n*cols_out + k] = in[k*rows_out + n]  (used for Wg: (512,256) -> N-major (256,512))
__global__ void k_transpose_h(const float* __restrict__ in, _Float16* __restrict__ out,
                              int rows_out, int cols_out) {
  int idx = blockIdx.x * blockDim.x + threadIdx.x;
  if (idx >= rows_out * cols_out) return;
  int n = idx / cols_out, k = idx % cols_out;
  out[idx] = (_Float16)in[k * rows_out + n];
}

__global__ void k_bias_sum(const float* __restrict__ a, const float* __restrict__ b,
                           float* __restrict__ o, int n) {
  int i = blockIdx.x * blockDim.x + threadIdx.x;
  if (i < n) o[i] = a[i] + b[i];
}

__global__ void k_zero_f16(_Float16* p, int n) {
  int i = blockIdx.x * blockDim.x + threadIdx.x;
  if (i < n) p[i] = (_Float16)0.0f;
}
__global__ void k_zero_f32(float* p, int n) {
  int i = blockIdx.x * blockDim.x + threadIdx.x;
  if (i < n) p[i] = 0.0f;
}
__global__ void k_f32_to_f16(const float* __restrict__ in, _Float16* __restrict__ out, int n) {
  int i = blockIdx.x * blockDim.x + threadIdx.x;
  if (i < n) out[i] = (_Float16)in[i];
}

// xh row r = s*B + b, padded 300 -> 320, f16
__global__ void k_embed(const int* __restrict__ words, const float* __restrict__ emb,
                        _Float16* __restrict__ xh) {
  int idx = blockIdx.x * blockDim.x + threadIdx.x;
  if (idx >= N_ * EMBP) return;
  int n = idx / EMBP, c = idx % EMBP;
  int b = n & (B_ - 1), s = n >> 7;
  if (c < EMBD) {
    int w = words[b * S_ + s];
    xh[idx] = (_Float16)emb[(size_t)w * EMBD + c];
  } else {
    xh[idx] = (_Float16)0.0f;
  }
}

// ---------------- WMMA GEMM, 2x4 register blocking (32x64 output per wave) ----------------
// C[M,Ncols] = A[M,K](f16) * Bw[Ncols,K](f16, N-major) + bias
// requires M % 32 == 0, Ncols % 64 == 0, K % 32 == 0

__global__ void k_gemm_wmma(const _Float16* __restrict__ A, const _Float16* __restrict__ Bw,
                            const float* __restrict__ bias, float* __restrict__ C,
                            int M, int Ncols, int K) {
  int lane = threadIdx.x & 31;
  int wave = blockIdx.x * (blockDim.x >> 5) + (threadIdx.x >> 5);
  int ntn = Ncols >> 6;                    // 64-wide N blocks
  int total = (M >> 5) * ntn;              // 32-high M blocks
  if (wave >= total) return;
  int tn = wave % ntn, tm = wave / ntn;
  int hf = lane >> 4, ln = lane & 15;

  const _Float16* ap = A + (size_t)(tm * 32 + ln) * K + hf * 16;
  const size_t arow16 = (size_t)16 * K;    // 16 A-rows
  const _Float16* bp = Bw + (size_t)(tn * 64 + ln) * K + hf * 16;

  v8f acc[2][4];
#pragma unroll
  for (int mi = 0; mi < 2; ++mi)
#pragma unroll
    for (int nt = 0; nt < 4; ++nt) acc[mi][nt] = (v8f){};

  for (int kk = 0; kk < K; kk += 32) {
    __builtin_prefetch(ap + kk + 128, 0, 0);              // global_prefetch_b8
    __builtin_prefetch(ap + arow16 + kk + 128, 0, 0);
    __builtin_prefetch(bp + kk + 256, 0, 0);

    union { v16h v; v8h h[2]; } ua[2];
#pragma unroll
    for (int mi = 0; mi < 2; ++mi) {
      const _Float16* p = ap + (size_t)mi * arow16 + kk;
      ua[mi].h[0] = *(const v8h*)(p);        // K = kk + hf*16 .. +7     -> elems 0..7
      ua[mi].h[1] = *(const v8h*)(p + 8);    // K = kk + hf*16 + 8 .. +7 -> elems 8..15
    }
    v16h bf[4];
#pragma unroll
    for (int nt = 0; nt < 4; ++nt)
      bf[nt] = *(const v16h*)(bp + (size_t)nt * arow16 + kk);  // 16 B-rows stride == 16*K

#pragma unroll
    for (int mi = 0; mi < 2; ++mi)
#pragma unroll
      for (int nt = 0; nt < 4; ++nt)
        acc[mi][nt] = __builtin_amdgcn_wmma_f32_16x16x32_f16(false, ua[mi].v, false, bf[nt],
                                                             (short)0, acc[mi][nt], false, false);
  }

#pragma unroll
  for (int mi = 0; mi < 2; ++mi) {
#pragma unroll
    for (int nt = 0; nt < 4; ++nt) {
      int n = tn * 64 + nt * 16 + ln;
      float bb = bias ? bias[n] : 0.0f;
      float* cp = C + (size_t)(tm * 32 + mi * 16 + 8 * hf) * Ncols + n;
#pragma unroll
      for (int v = 0; v < 8; ++v) cp[(size_t)v * Ncols] = acc[mi][nt][v] + bb;
    }
  }
}

// ---------------- fused LSTM step: both directions, h@Whh.T + gates + cell ----------------

__global__ void k_lstm_step(const float* __restrict__ gf, const float* __restrict__ gb,
                            const _Float16* __restrict__ Wf, const _Float16* __restrict__ Wb,
                            const _Float16* __restrict__ hpf, const _Float16* __restrict__ hpb,
                            _Float16* __restrict__ hnf, _Float16* __restrict__ hnb,
                            float* __restrict__ cf, float* __restrict__ cb,
                            _Float16* __restrict__ out, int swap_rs, int t) {
  int lane = threadIdx.x & 31;
  int wave = blockIdx.x * 8 + (threadIdx.x >> 5);   // 32 blocks * 8 waves = 256 waves
  int dir = wave >> 7;                              // 0 = fwd, 1 = bwd
  int rem = wave & 127;
  int tm = rem & 7, tj = rem >> 3;                  // 8 M-tiles x 16 J-tiles
  int s = dir ? (S_ - 1 - t) : t;
  const float*    gp = dir ? gb : gf;
  const _Float16* W  = dir ? Wb : Wf;
  const _Float16* hp = dir ? hpb : hpf;
  _Float16*       hn = dir ? hnb : hnf;
  float*          cc = dir ? cb : cf;

  int hf = lane >> 4, ln = lane & 15;
  const _Float16* ap = hp + (tm * 16 + ln) * H_ + hf * 16;
  v8f acc[4] = {{}, {}, {}, {}};
#pragma unroll
  for (int kk = 0; kk < H_; kk += 32) {
    union { v16h v; v8h h[2]; } ua;
    ua.h[0] = *(const v8h*)(ap + kk);
    ua.h[1] = *(const v8h*)(ap + kk + 8);
#pragma unroll
    for (int g = 0; g < 4; ++g) {
      const _Float16* bp = W + (size_t)(g * 256 + tj * 16 + ln) * H_ + hf * 16 + kk;
      v16h bf = *(const v16h*)bp;
      acc[g] = __builtin_amdgcn_wmma_f32_16x16x32_f16(false, ua.v, false, bf,
                                                      (short)0, acc[g], false, false);
    }
  }
  int j = tj * 16 + ln;
#pragma unroll
  for (int v = 0; v < 8; ++v) {
    int bb = tm * 16 + v + 8 * hf;                  // batch row
    size_t grow = ((size_t)s * B_ + bb) * (size_t)G4;
    float ig = acc[0][v] + gp[grow + j];
    float fg = acc[1][v] + gp[grow + 256 + j];
    float gg = acc[2][v] + gp[grow + 512 + j];
    float og = acc[3][v] + gp[grow + 768 + j];
    float co = cc[bb * H_ + j];
    float cn = sigf(fg) * co + sigf(ig) * tanhf(gg);
    float hv = sigf(og) * tanhf(cn);
    cc[bb * H_ + j] = cn;
    _Float16 hh = (_Float16)hv;
    hn[bb * H_ + j] = hh;
    size_t orow = swap_rs ? ((size_t)bb * S_ + s) : ((size_t)s * B_ + bb);
    out[orow * 512 + (size_t)dir * 256 + j] = hh;
  }
}

// ---------------- GCN ----------------

__global__ void k_deg_init(float* deg) {
  int i = blockIdx.x * blockDim.x + threadIdx.x;
  if (i < N_) deg[i] = 1.0f;  // self loop
}
__global__ void k_deg_edges(const int* __restrict__ col, float* __restrict__ deg) {
  int e = blockIdx.x * blockDim.x + threadIdx.x;
  if (e < NE_) atomicAdd(&deg[col[e]], 1.0f);
}
__global__ void k_rsqrt(float* d) {
  int i = blockIdx.x * blockDim.x + threadIdx.x;
  if (i < N_) d[i] = rsqrtf(d[i]);
}
__global__ void k_agg_init(const float* __restrict__ xw, const float* __restrict__ dinv,
                           const float* __restrict__ bg, float* __restrict__ agg) {
  int idx = blockIdx.x * blockDim.x + threadIdx.x;
  if (idx >= N_ * H_) return;
  int n = idx >> 8, j = idx & 255;
  float di = dinv[n];
  agg[idx] = xw[idx] * di * di + bg[j];   // self-loop term + bias
}
__global__ void k_agg_edges(const int* __restrict__ row, const int* __restrict__ col,
                            const float* __restrict__ xw, const float* __restrict__ dinv,
                            float* __restrict__ agg) {
  int e = blockIdx.x;
  int j = threadIdx.x;
  int r = row[e], c = col[e];
  float w = dinv[r] * dinv[c];
  atomicAdd(&agg[(size_t)c * H_ + j], xw[(size_t)r * H_ + j] * w);
}

// ---------------- attention ----------------

__global__ void k_qp(const float* __restrict__ gout, const float* __restrict__ Wv,
                     float* __restrict__ qp) {
  int b = blockIdx.x, a = threadIdx.x;
  const float* q = gout + (size_t)b * S_ * H_;   // row (b, s=0)
  float acc = 0.0f;
  for (int h = 0; h < H_; ++h) acc += Wv[a * H_ + h] * q[h];
  qp[b * H_ + a] = acc;
}

__global__ void k_scores(const float* __restrict__ t1, const float* __restrict__ qp,
                         const float* __restrict__ Wt, const float* __restrict__ bt,
                         const unsigned char* __restrict__ smask,
                         const unsigned char* __restrict__ omask,
                         float* __restrict__ sw, float* __restrict__ ow) {
  int b = blockIdx.x, s = threadIdx.x;
  __shared__ float red[256];
  const float* tp = t1 + ((size_t)b * S_ + s) * H_;
  const float* qb = qp + b * H_;
  float sc = 0.0f;
  for (int a = 0; a < H_; ++a) sc += Wt[a] * tanhf(tp[a] + qb[a]);
  sc += bt[0];

  // subj softmax
  {
    float x = smask[b * S_ + s] ? -__builtin_inff() : sc;
    red[s] = x; __syncthreads();
    for (int o = 128; o > 0; o >>= 1) { if (s < o) red[s] = fmaxf(red[s], red[s + o]); __syncthreads(); }
    float mx = red[0]; __syncthreads();
    float e = __expf(x - mx);
    red[s] = e; __syncthreads();
    for (int o = 128; o > 0; o >>= 1) { if (s < o) red[s] += red[s + o]; __syncthreads(); }
    float sm = red[0]; __syncthreads();
    sw[b * S_ + s] = e / sm;
  }
  // obj softmax
  {
    float x = omask[b * S_ + s] ? -__builtin_inff() : sc;
    red[s] = x; __syncthreads();
    for (int o = 128; o > 0; o >>= 1) { if (s < o) red[s] = fmaxf(red[s], red[s + o]); __syncthreads(); }
    float mx = red[0]; __syncthreads();
    float e = __expf(x - mx);
    red[s] = e; __syncthreads();
    for (int o = 128; o > 0; o >>= 1) { if (s < o) red[s] += red[s + o]; __syncthreads(); }
    float sm = red[0]; __syncthreads();
    ow[b * S_ + s] = e / sm;
  }
}

__global__ void k_pool(const float* __restrict__ sw, const float* __restrict__ ow,
                       const float* __restrict__ gout, float* __restrict__ fin) {
  int b = blockIdx.x, h = threadIdx.x;
  float s1 = 0.0f, s2 = 0.0f;
  for (int s = 0; s < S_; ++s) {
    float g = gout[((size_t)b * S_ + s) * H_ + h];
    s1 += sw[b * S_ + s] * g;
    s2 += ow[b * S_ + s] * g;
  }
  fin[b * 512 + h] = s1;
  fin[b * 512 + 256 + h] = s2;
}

__global__ void k_logits(const float* __restrict__ fin, const float* __restrict__ Wl,
                         const float* __restrict__ bl, float* __restrict__ out) {
  int b = blockIdx.x, c = threadIdx.x;
  if (c >= NC_) return;
  float acc = bl[c];
  const float* f = fin + b * 512;
  const float* w = Wl + c * 512;
  for (int k = 0; k < 512; ++k) acc += f[k] * w[k];
  out[b * NC_ + c] = acc;
}

// ---------------- host ----------------

extern "C" void kernel_launch(void* const* d_in, const int* in_sizes, int n_in,
                              void* d_out, int out_size, void* d_ws, size_t ws_size,
                              hipStream_t stream) {
  (void)in_sizes; (void)n_in; (void)out_size; (void)ws_size;
  const int*           words = (const int*)d_in[0];
  const unsigned char* smask = (const unsigned char*)d_in[2];
  const unsigned char* omask = (const unsigned char*)d_in[3];
  const int*           ei    = (const int*)d_in[4];   // (2, NE): row = ei, col = ei+NE
  const float* emb   = (const float*)d_in[5];
  const float* Wih0f = (const float*)d_in[6];
  const float* Whh0f = (const float*)d_in[7];
  const float* bih0f = (const float*)d_in[8];
  const float* bhh0f = (const float*)d_in[9];
  const float* Wih0b = (const float*)d_in[10];
  const float* Whh0b = (const float*)d_in[11];
  const float* bih0b = (const float*)d_in[12];
  const float* bhh0b = (const float*)d_in[13];
  const float* Wih1f = (const float*)d_in[14];
  const float* Whh1f = (const float*)d_in[15];
  const float* bih1f = (const float*)d_in[16];
  const float* bhh1f = (const float*)d_in[17];
  const float* Wih1b = (const float*)d_in[18];
  const float* Whh1b = (const float*)d_in[19];
  const float* bih1b = (const float*)d_in[20];
  const float* bhh1b = (const float*)d_in[21];
  const float* Wg = (const float*)d_in[22];
  const float* bg = (const float*)d_in[23];
  const float* Wu = (const float*)d_in[24];
  const float* bu = (const float*)d_in[25];
  const float* Wv = (const float*)d_in[26];
  const float* Wt = (const float*)d_in[27];
  const float* bt = (const float*)d_in[28];
  const float* Wl = (const float*)d_in[29];
  const float* bl = (const float*)d_in[30];

  char* base = (char*)d_ws;
  size_t off = 0;
  auto take = [&](size_t bytes) -> char* {
    char* p = base + off;
    off += (bytes + 255) & ~(size_t)255;
    return p;
  };

  _Float16* xh      = (_Float16*)take((size_t)N_ * EMBP * 2);
  _Float16* Wih0f_h = (_Float16*)take((size_t)G4 * EMBP * 2);
  _Float16* Wih0b_h = (_Float16*)take((size_t)G4 * EMBP * 2);
  _Float16* Whh0f_h = (_Float16*)take((size_t)G4 * H_ * 2);
  _Float16* Whh0b_h = (_Float16*)take((size_t)G4 * H_ * 2);
  _Float16* Wih1f_h = (_Float16*)take((size_t)G4 * 512 * 2);
  _Float16* Wih1b_h = (_Float16*)take((size_t)G4 * 512 * 2);
  _Float16* Whh1f_h = (_Float16*)take((size_t)G4 * H_ * 2);
  _Float16* Whh1b_h = (_Float16*)take((size_t)G4 * H_ * 2);
  _Float16* WgT_h   = (_Float16*)take((size_t)H_ * 512 * 2);
  _Float16* Wu_h    = (_Float16*)take((size_t)H_ * H_ * 2);
  float* bias0f = (float*)take(G4 * 4);
  float* bias0b = (float*)take(G4 * 4);
  float* bias1f = (float*)take(G4 * 4);
  float* bias1b = (float*)take(G4 * 4);
  float* g_f = (float*)take((size_t)N_ * G4 * 4);   // reused: layer0 fwd gates -> layer1 fwd gates -> GCN/attn scratch
  float* g_b = (float*)take((size_t)N_ * G4 * 4);
  _Float16* h0_h    = (_Float16*)take((size_t)N_ * 512 * 2);
  _Float16* nodes_h = (_Float16*)take((size_t)N_ * 512 * 2);
  _Float16* hb[2][2];
  for (int d = 0; d < 2; ++d)
    for (int p = 0; p < 2; ++p) hb[d][p] = (_Float16*)take((size_t)B_ * H_ * 2);
  float* cbf = (float*)take((size_t)B_ * H_ * 4);
  float* cbb = (float*)take((size_t)B_ * H_ * 4);
  float* deg = (float*)take((size_t)N_ * 4);
  float* qp  = (float*)take((size_t)B_ * H_ * 4);
  float* sw  = (float*)take((size_t)B_ * S_ * 4);
  float* ow  = (float*)take((size_t)B_ * S_ * 4);
  float* fin = (float*)take((size_t)B_ * 512 * 4);

  // overlays inside g_f (free after layer-1 recurrence): all fit in 134 MB
  float*    xw     = (float*)((char*)g_f + 0);
  float*    agg    = (float*)((char*)g_f + ((size_t)36 << 20));
  float*    t1     = (float*)((char*)g_f + ((size_t)72 << 20));
  _Float16* gout_h = (_Float16*)((char*)g_f + ((size_t)108 << 20));

  auto grid1 = [](int n) { return dim3((unsigned)((n + 255) / 256)); };
  auto gemm = [&](const _Float16* Aa, const _Float16* Bb, const float* bias, float* Cc,
                  int M, int Nc, int K) {
    int waves = (M >> 5) * (Nc >> 6);     // 32x64 block per wave
    int blocks = (waves + 7) >> 3;
    k_gemm_wmma<<<blocks, 256, 0, stream>>>(Aa, Bb, bias, Cc, M, Nc, K);
  };

  // ---- weight prep ----
  k_convert_pad<<<grid1(G4 * EMBP), 256, 0, stream>>>(Wih0f, Wih0f_h, G4, EMBD, EMBP);
  k_convert_pad<<<grid1(G4 * EMBP), 256, 0, stream>>>(Wih0b, Wih0b_h, G4, EMBD, EMBP);
  k_convert_pad<<<grid1(G4 * H_), 256, 0, stream>>>(Whh0f, Whh0f_h, G4, H_, H_);
  k_convert_pad<<<grid1(G4 * H_), 256, 0, stream>>>(Whh0b, Whh0b_h, G4, H_, H_);
  k_convert_pad<<<grid1(G4 * 512), 256, 0, stream>>>(Wih1f, Wih1f_h, G4, 512, 512);
  k_convert_pad<<<grid1(G4 * 512), 256, 0, stream>>>(Wih1b, Wih1b_h, G4, 512, 512);
  k_convert_pad<<<grid1(G4 * H_), 256, 0, stream>>>(Whh1f, Whh1f_h, G4, H_, H_);
  k_convert_pad<<<grid1(G4 * H_), 256, 0, stream>>>(Whh1b, Whh1b_h, G4, H_, H_);
  k_transpose_h<<<grid1(H_ * 512), 256, 0, stream>>>(Wg, WgT_h, H_, 512);
  k_convert_pad<<<grid1(H_ * H_), 256, 0, stream>>>(Wu, Wu_h, H_, H_, H_);
  k_bias_sum<<<grid1(G4), 256, 0, stream>>>(bih0f, bhh0f, bias0f, G4);
  k_bias_sum<<<grid1(G4), 256, 0, stream>>>(bih0b, bhh0b, bias0b, G4);
  k_bias_sum<<<grid1(G4), 256, 0, stream>>>(bih1f, bhh1f, bias1f, G4);
  k_bias_sum<<<grid1(G4), 256, 0, stream>>>(bih1b, bhh1b, bias1b, G4);

  // ---- embedding gather (s,b order) + layer-0 input projections ----
  k_embed<<<grid1(N_ * EMBP), 256, 0, stream>>>(words, emb, xh);
  gemm(xh, Wih0f_h, bias0f, g_f, N_, G4, EMBP);
  gemm(xh, Wih0b_h, bias0b, g_b, N_, G4, EMBP);

  // ---- layer-0 recurrence (both directions per step) ----
  k_zero_f16<<<grid1(B_ * H_), 256, 0, stream>>>(hb[0][0], B_ * H_);
  k_zero_f16<<<grid1(B_ * H_), 256, 0, stream>>>(hb[1][0], B_ * H_);
  k_zero_f32<<<grid1(B_ * H_), 256, 0, stream>>>(cbf, B_ * H_);
  k_zero_f32<<<grid1(B_ * H_), 256, 0, stream>>>(cbb, B_ * H_);
  for (int t = 0; t < S_; ++t) {
    int p = t & 1;
    k_lstm_step<<<32, 256, 0, stream>>>(g_f, g_b, Whh0f_h, Whh0b_h,
                                        hb[0][p], hb[1][p], hb[0][p ^ 1], hb[1][p ^ 1],
                                        cbf, cbb, h0_h, 0, t);
  }

  // ---- layer-1 input projections + recurrence (outputs in (b,s) node order) ----
  gemm(h0_h, Wih1f_h, bias1f, g_f, N_, G4, 512);
  gemm(h0_h, Wih1b_h, bias1b, g_b, N_, G4, 512);
  k_zero_f16<<<grid1(B_ * H_), 256, 0, stream>>>(hb[0][0], B_ * H_);
  k_zero_f16<<<grid1(B_ * H_), 256, 0, stream>>>(hb[1][0], B_ * H_);
  k_zero_f32<<<grid1(B_ * H_), 256, 0, stream>>>(cbf, B_ * H_);
  k_zero_f32<<<grid1(B_ * H_), 256, 0, stream>>>(cbb, B_ * H_);
  for (int t = 0; t < S_; ++t) {
    int p = t & 1;
    k_lstm_step<<<32, 256, 0, stream>>>(g_f, g_b, Whh1f_h, Whh1b_h,
                                        hb[0][p], hb[1][p], hb[0][p ^ 1], hb[1][p ^ 1],
                                        cbf, cbb, nodes_h, 1, t);
  }

  // ---- GCN: xw = nodes @ Wg ; normalized segment-sum with self loops ; + bg ----
  gemm(nodes_h, WgT_h, nullptr, xw, N_, H_, 512);
  k_deg_init<<<grid1(N_), 256, 0, stream>>>(deg);
  k_deg_edges<<<grid1(NE_), 256, 0, stream>>>(ei + NE_, deg);
  k_rsqrt<<<grid1(N_), 256, 0, stream>>>(deg);            // deg -> dinv in place
  k_agg_init<<<grid1(N_ * H_), 256, 0, stream>>>(xw, deg, bg, agg);
  k_agg_edges<<<NE_, 256, 0, stream>>>(ei, ei + NE_, xw, deg, agg);

  // ---- attention: t1 = gout @ Wu.T + bu (WMMA), qp = q @ Wv.T, scores, softmax, pool ----
  k_f32_to_f16<<<grid1(N_ * H_), 256, 0, stream>>>(agg, gout_h, N_ * H_);
  gemm(gout_h, Wu_h, bu, t1, N_, H_, H_);
  k_qp<<<B_, H_, 0, stream>>>(agg, Wv, qp);
  k_scores<<<B_, S_, 0, stream>>>(t1, qp, Wt, bt, smask, omask, sw, ow);
  k_pool<<<B_, H_, 0, stream>>>(sw, ow, agg, fin);

  // ---- classifier ----
  k_logits<<<B_, 64, 0, stream>>>(fin, Wl, bl, (float*)d_out);
}
